// dvhNet_13091060318687
// MI455X (gfx1250) — matrix operand order repaired and checked by reading, not compiled, not run on hardware
//
#include <hip/hip_runtime.h>

typedef __attribute__((ext_vector_type(16))) _Float16 v16h;
typedef __attribute__((ext_vector_type(2)))  _Float16 v2h;
typedef __attribute__((ext_vector_type(8)))  float    v8f;

#define HDIM  256
#define CDIM  256
#define TLEN  16384
#define BATCH 8
#define NBLK  5
#define NTILE 64              // t-columns per workgroup tile
#define EPS_BN 1e-5f
#define NTOT  (BATCH * TLEN)  // reduction count for BN stats (131072)

// ---------------------------------------------------------------------------
// zero both stats buffers (1024 floats)
__global__ void zero_stats_kernel(float* sA, float* sB) {
    int t = threadIdx.x;
    sA[t] = 0.f; sA[t + 256] = 0.f;
    sB[t] = 0.f; sB[t + 256] = 0.f;
}

// ---------------------------------------------------------------------------
// Pack the 10 H x H f32 weight matrices (fc0_W[0..4], fc1_W[0..4] interleaved
// as layer = 2i / 2i+1) into f16 WMMA A-fragment order:
//   Wpk[layer][mt][ks][lane][j],  j contiguous -> one 32B load per lane/frag.
// A 16x32 f16 fragment as packed here (B uses the same k order): M = lane%16, K = j + 16*(lane/16).
__global__ __launch_bounds__(256) void pack_w_kernel(
        const float* __restrict__ fc0W, const float* __restrict__ fc1W,
        _Float16* __restrict__ Wpk) {
    int flat  = blockIdx.x * 256 + threadIdx.x;       // < 10*65536
    int j     = flat & 15;
    int lane  = (flat >> 4) & 31;
    int ks    = (flat >> 9) & 7;
    int mt    = (flat >> 12) & 15;
    int layer = flat >> 16;
    int m     = lane & 15;
    int khalf = lane >> 4;
    int k     = j + khalf * 16;
    int oc    = mt * 16 + m;
    int ic    = ks * 32 + k;
    const float* src = (layer & 1) ? fc1W : fc0W;
    float v = src[(size_t)(layer >> 1) * HDIM * HDIM + (size_t)oc * HDIM + ic];
    Wpk[flat] = (_Float16)v;
}

// ---------------------------------------------------------------------------
// gamma/beta = c @ W^T + b   for one BN use.  grid(BATCH), block(HDIM)
__global__ __launch_bounds__(256) void affine_kernel(
        const float* __restrict__ cvec,
        const float* __restrict__ Wg, const float* __restrict__ bg,
        const float* __restrict__ Wb, const float* __restrict__ bb,
        float* __restrict__ gamma, float* __restrict__ beta) {
    __shared__ float csh[CDIM];
    int b = blockIdx.x, h = threadIdx.x;
    csh[h] = cvec[b * CDIM + h];
    __syncthreads();
    float g = 0.f, bt = 0.f;
    const float* wgr = Wg + (size_t)h * CDIM;
    const float* wbr = Wb + (size_t)h * CDIM;
    for (int k = 0; k < CDIM; ++k) {
        float cv = csh[k];
        g  += wgr[k] * cv;
        bt += wbr[k] * cv;
    }
    gamma[b * HDIM + h] = g + bg[h];
    beta [b * HDIM + h] = bt + bb[h];
}

// ---------------------------------------------------------------------------
// stats(sum,sumsq) + gamma/beta -> per (b,c) affine  act = relu(A*x + B).
// Also zeros the stats buffer that the NEXT producer will accumulate into.
// grid(BATCH), block(HDIM)
__global__ __launch_bounds__(256) void finalize_kernel(
        const float* __restrict__ stats,
        const float* __restrict__ gamma, const float* __restrict__ beta,
        float* __restrict__ coef, float* __restrict__ nextStats) {
    int b = blockIdx.x, ch = threadIdx.x;
    float mean = stats[ch] * (1.0f / NTOT);
    float var  = stats[HDIM + ch] * (1.0f / NTOT) - mean * mean;
    float rs   = rsqrtf(var + EPS_BN);
    float g    = gamma[b * HDIM + ch];
    float A    = g * rs;
    coef[b * HDIM + ch] = A;
    coef[BATCH * HDIM + b * HDIM + ch] = beta[b * HDIM + ch] - A * mean;
    if (b == 0) { nextStats[ch] = 0.f; nextStats[HDIM + ch] = 0.f; }
}

// ---------------------------------------------------------------------------
// fc_p: x[b,h,t] = W[h,:]·p[b,t,:] + b[h];  fused per-channel stats (atomics)
// grid(BATCH * TLEN/NTILE), block(256)
__global__ __launch_bounds__(256) void fcp_kernel(
        const float* __restrict__ pts, const float* __restrict__ W,
        const float* __restrict__ bias, float* __restrict__ x,
        float* __restrict__ stats) {
    __shared__ float psh[NTILE * 3];
    __shared__ float wsh[HDIM * 3];
    __shared__ float bsh[HDIM];
    int b  = blockIdx.x >> 8;
    int t0 = (blockIdx.x & 255) * NTILE;
    int tid = threadIdx.x;
    if (tid < NTILE * 3) psh[tid] = pts[(size_t)(b * TLEN + t0) * 3 + tid];
    for (int i = tid; i < HDIM * 3; i += 256) wsh[i] = W[i];
    bsh[tid] = bias[tid];
    __syncthreads();
    int t    = tid & 63;
    int cgrp = tid >> 6;
    float px = psh[t * 3 + 0], py = psh[t * 3 + 1], pz = psh[t * 3 + 2];
    for (int i = 0; i < 64; ++i) {
        int ch = cgrp * 64 + i;
        float v = wsh[ch * 3] * px + wsh[ch * 3 + 1] * py + wsh[ch * 3 + 2] * pz + bsh[ch];
        x[(size_t)(b * HDIM + ch) * TLEN + t0 + t] = v;
        float s1 = v, s2 = v * v;
        for (int m = 1; m < 32; m <<= 1) {
            s1 += __shfl_xor(s1, m, 32);
            s2 += __shfl_xor(s2, m, 32);
        }
        if ((tid & 31) == 0) {
            atomicAdd(&stats[ch], s1);
            atomicAdd(&stats[HDIM + ch], s2);
        }
    }
}

// ---------------------------------------------------------------------------
// Fused CBN(affine)+ReLU -> f16 -> WMMA GEMM (256x64 tile) -> bias
// (+ optional residual RMW on dst) -> fused per-channel stats of the output.
// grid(BATCH * TLEN/NTILE), block(256) = 8 waves; each wave owns 32 out-chans.
__global__ __launch_bounds__(256) void gemm_cbn_kernel(
        const float* __restrict__ src, float* __restrict__ dst,
        const _Float16* __restrict__ Wpk, const float* __restrict__ bias,
        const float* __restrict__ coef, float* __restrict__ stats,
        int residual) {
    __shared__ __align__(32) _Float16 alds[16384];   // 32 KB: packed B frags
    const int tid = threadIdx.x;
    const int b   = blockIdx.x >> 8;
    const int t0  = (blockIdx.x & 255) * NTILE;

    // ---- stage: normalize+ReLU+cvt f16, packed into B-fragment lane order.
    // B 32x16 f16 fragment as packed here (same k order as Wpk): col n = lane%16, K = j + 16*(lane/16).
    {
        const int c0   = (tid >> 1) * 2;   // even channel; thread covers c0,c0+1
        const int half = tid & 1;
        const int tb   = half * 32;        // 32 t-columns per thread
        const float A0 = coef[b * HDIM + c0];
        const float B0 = coef[BATCH * HDIM + b * HDIM + c0];
        const float A1 = coef[b * HDIM + c0 + 1];
        const float B1 = coef[BATCH * HDIM + b * HDIM + c0 + 1];
        const float4* r0 = (const float4*)(src + (size_t)(b * HDIM + c0) * TLEN + t0 + tb);
        const float4* r1 = (const float4*)(src + (size_t)(b * HDIM + c0 + 1) * TLEN + t0 + tb);
        const int kstep = c0 >> 5;
        const int kin   = c0 & 31;
        const int khalf = kin >> 4;
        const int j     = kin & 15;        // even -> pair store is 4B aligned
        for (int q = 0; q < 8; ++q) {
            float4 v0 = r0[q];
            float4 v1 = r1[q];
            float e0[4] = {v0.x, v0.y, v0.z, v0.w};
            float e1[4] = {v1.x, v1.y, v1.z, v1.w};
            for (int e = 0; e < 4; ++e) {
                int n    = tb + q * 4 + e;
                int nt   = n >> 4;
                int lane = khalf * 16 + (n & 15);
                int base = ((kstep * 4 + nt) * 32 + lane) * 16 + j;
                v2h pr;
                pr[0] = (_Float16)fmaxf(A0 * e0[e] + B0, 0.f);
                pr[1] = (_Float16)fmaxf(A1 * e1[e] + B1, 0.f);
                *(v2h*)&alds[base] = pr;
            }
        }
    }
    __syncthreads();

    const int wave = tid >> 5;
    const int lane = tid & 31;

    v8f acc[2][4];
    for (int mi = 0; mi < 2; ++mi)
        for (int nt = 0; nt < 4; ++nt) { v8f z = {}; acc[mi][nt] = z; }

    for (int ks = 0; ks < 8; ++ks) {
        v16h af[2], bf[4];
        for (int mi = 0; mi < 2; ++mi) {
            int mt = wave * 2 + mi;
            af[mi] = *(const v16h*)(Wpk + (size_t)(((mt * 8 + ks) * 32 + lane) << 4));
        }
        for (int nt = 0; nt < 4; ++nt)
            bf[nt] = *(const v16h*)(alds + (((ks * 4 + nt) * 32 + lane) << 4));
        for (int mi = 0; mi < 2; ++mi)
            for (int nt = 0; nt < 4; ++nt)
                acc[mi][nt] = __builtin_amdgcn_wmma_f32_16x16x32_f16(
                    false, af[mi], false, bf[nt], (short)0, acc[mi][nt], false, false);
    }

    // ---- epilogue: bias (+residual) store, in-wave stats reduce, atomics.
    // D f32 16x16: row m = v + 8*(lane/16), col n = lane%16.
    const int khalf = lane >> 4;
    const int ncol  = lane & 15;
    for (int mi = 0; mi < 2; ++mi) {
        const int mt = wave * 2 + mi;
        float s1[8], s2[8];
        for (int v = 0; v < 8; ++v) {
            const int ch = mt * 16 + v + khalf * 8;
            const float bv = bias[ch];
            float* drow = dst + (size_t)(b * HDIM + ch) * TLEN + t0;
            float a1 = 0.f, a2 = 0.f;
            for (int nt = 0; nt < 4; ++nt) {
                int tt = nt * 16 + ncol;
                float val = acc[mi][nt][v] + bv;
                if (residual) val += drow[tt];
                drow[tt] = val;
                a1 += val; a2 += val * val;
            }
            s1[v] = a1; s2[v] = a2;
        }
        for (int v = 0; v < 8; ++v)
            for (int m = 1; m < 16; m <<= 1) {   // reduce across 16-lane group
                s1[v] += __shfl_xor(s1[v], m, 32);
                s2[v] += __shfl_xor(s2[v], m, 32);
            }
        if (ncol == 0) {
            for (int v = 0; v < 8; ++v) {
                const int ch = mt * 16 + v + khalf * 8;
                atomicAdd(&stats[ch], s1[v]);
                atomicAdd(&stats[HDIM + ch], s2[v]);
            }
        }
    }
}

// ---------------------------------------------------------------------------
// final: out[b,t] = sum_c out_W[c]*relu(A*x + B) + out_b
// grid(BATCH * TLEN/256), block(256)
__global__ __launch_bounds__(256) void out_kernel(
        const float* __restrict__ x, const float* __restrict__ coef,
        const float* __restrict__ outW, const float* __restrict__ outB,
        float* __restrict__ out) {
    __shared__ float wsh[HDIM];
    int b = blockIdx.x >> 6;
    int t = ((blockIdx.x & 63) << 8) + threadIdx.x;
    wsh[threadIdx.x] = outW[threadIdx.x];
    __syncthreads();
    const float* cA = coef + b * HDIM;
    const float* cB = coef + BATCH * HDIM + b * HDIM;
    float acc = 0.f;
    for (int c = 0; c < HDIM; ++c) {
        float v = x[(size_t)(b * HDIM + c) * TLEN + t];
        acc += wsh[c] * fmaxf(cA[c] * v + cB[c], 0.f);
    }
    out[b * TLEN + t] = acc + outB[0];
}

// ---------------------------------------------------------------------------
extern "C" void kernel_launch(void* const* d_in, const int* in_sizes, int n_in,
                              void* d_out, int out_size, void* d_ws, size_t ws_size,
                              hipStream_t stream) {
    const float* points = (const float*)d_in[0];
    const float* cvec   = (const float*)d_in[1];
    const float* fc_p_W = (const float*)d_in[2];
    const float* fc_p_b = (const float*)d_in[3];
    const float* bn0_Wg = (const float*)d_in[4];
    const float* bn0_bg = (const float*)d_in[5];
    const float* bn0_Wb = (const float*)d_in[6];
    const float* bn0_bb = (const float*)d_in[7];
    const float* bn1_Wg = (const float*)d_in[8];
    const float* bn1_bg = (const float*)d_in[9];
    const float* bn1_Wb = (const float*)d_in[10];
    const float* bn1_bb = (const float*)d_in[11];
    const float* fc0_W  = (const float*)d_in[12];
    const float* fc0_b  = (const float*)d_in[13];
    const float* fc1_W  = (const float*)d_in[14];
    const float* fc1_b  = (const float*)d_in[15];
    const float* bnf_Wg = (const float*)d_in[16];
    const float* bnf_bg = (const float*)d_in[17];
    const float* bnf_Wb = (const float*)d_in[18];
    const float* bnf_bb = (const float*)d_in[19];
    const float* out_W  = (const float*)d_in[20];
    const float* out_b  = (const float*)d_in[21];
    float* outp = (float*)d_out;

    const size_t tens = (size_t)BATCH * HDIM * TLEN * 4;   // 134,217,728 B
    char* ws = (char*)d_ws;
    size_t off = 0;
    float*     x     = (float*)(ws + off);  off += tens;
    float*     net   = (float*)(ws + off);  off += tens;
    _Float16*  Wpk   = (_Float16*)(ws + off); off += (size_t)10 * HDIM * HDIM * 2;
    float*     gb    = (float*)(ws + off);  off += (size_t)11 * 2 * BATCH * HDIM * 4;
    float*     coef0 = (float*)(ws + off);  off += (size_t)2 * BATCH * HDIM * 4;
    float*     coef1 = (float*)(ws + off);  off += (size_t)2 * BATCH * HDIM * 4;
    float*     stA   = (float*)(ws + off);  off += 2 * HDIM * 4;
    float*     stB   = (float*)(ws + off);  off += 2 * HDIM * 4;
    if (off > ws_size) return;  // workspace too small; nothing safe to do

    const int GB_STRIDE = 2 * BATCH * HDIM;            // per BN-use
    auto GAMMA = [&](int u) { return gb + (size_t)u * GB_STRIDE; };
    auto BETA  = [&](int u) { return gb + (size_t)u * GB_STRIDE + BATCH * HDIM; };

    zero_stats_kernel<<<1, 256, 0, stream>>>(stA, stB);
    pack_w_kernel<<<2560, 256, 0, stream>>>(fc0_W, fc1_W, Wpk);

    for (int i = 0; i < NBLK; ++i) {
        affine_kernel<<<BATCH, 256, 0, stream>>>(cvec,
            bn0_Wg + (size_t)i * HDIM * CDIM, bn0_bg + i * HDIM,
            bn0_Wb + (size_t)i * HDIM * CDIM, bn0_bb + i * HDIM,
            GAMMA(i), BETA(i));
        affine_kernel<<<BATCH, 256, 0, stream>>>(cvec,
            bn1_Wg + (size_t)i * HDIM * CDIM, bn1_bg + i * HDIM,
            bn1_Wb + (size_t)i * HDIM * CDIM, bn1_bb + i * HDIM,
            GAMMA(5 + i), BETA(5 + i));
    }
    affine_kernel<<<BATCH, 256, 0, stream>>>(cvec, bnf_Wg, bnf_bg, bnf_Wb, bnf_bb,
                                             GAMMA(10), BETA(10));

    const int NWG = BATCH * (TLEN / NTILE);            // 2048
    fcp_kernel<<<NWG, 256, 0, stream>>>(points, fc_p_W, fc_p_b, x, stA);

    for (int i = 0; i < NBLK; ++i) {
        finalize_kernel<<<BATCH, 256, 0, stream>>>(stA, GAMMA(i), BETA(i), coef0, stB);
        gemm_cbn_kernel<<<NWG, 256, 0, stream>>>(
            x, net, Wpk + (size_t)(2 * i) * HDIM * HDIM, fc0_b + i * HDIM,
            coef0, stB, 0);
        finalize_kernel<<<BATCH, 256, 0, stream>>>(stB, GAMMA(5 + i), BETA(5 + i), coef1, stA);
        gemm_cbn_kernel<<<NWG, 256, 0, stream>>>(
            net, x, Wpk + (size_t)(2 * i + 1) * HDIM * HDIM, fc1_b + i * HDIM,
            coef1, stA, 1);
    }

    finalize_kernel<<<BATCH, 256, 0, stream>>>(stA, GAMMA(10), BETA(10), coef0, stB);
    out_kernel<<<BATCH * (TLEN / 256), 256, 0, stream>>>(x, coef0, out_W, out_b, outp);
    (void)in_sizes; (void)n_in; (void)out_size;
}
